// ScaledDotProductAttention_60249801228968
// MI455X (gfx1250) — matrix operand order repaired and checked by reading, not compiled, not run on hardware
//
#include <hip/hip_runtime.h>

// CDNA5 / gfx1250 fused attention:  out = softmax(Q K^T / scale) V, also emits attention.
// B=2 H=16 S=2048 D=64, fp32 in/out, f16 WMMA compute with f32 accumulation.
// K tiles staged into LDS via Tensor Data Mover (double-buffered, TENSORcnt),
// attention written as wide b128 stores (store-bandwidth-bound kernel).

typedef __attribute__((ext_vector_type(16))) _Float16 v16h;
typedef __attribute__((ext_vector_type(8)))  float    v8f;
typedef __attribute__((ext_vector_type(4)))  unsigned int u32x4;
typedef __attribute__((ext_vector_type(8)))  int      i32x8;
typedef __attribute__((ext_vector_type(4)))  int      i32x4;

#define B_  2
#define H_  16
#define S_  2048
#define D_  64
#define TILE_M 16
#define WAVES_PER_BLOCK 2

#define SC_FLOATS (TILE_M * S_)            // 32768 floats of score buffer per wave
#define KSTRIDE   65                       // staged K row stride in dwords (64 + 1 pad -> no bank conflicts)
#define KBUF_FLOATS (TILE_M * KSTRIDE)     // 1040 floats per staging buffer
#define NKBUF 2

// ---- TDM: stage one 16x64 f32 K-tile (rows are tensor dim1, cols dim0) into LDS ----
// D# per CDNA5 ISA: group0 = {flags, lds_addr, global_addr, type},
// group1 = {data_size/pad, tensor dims, tile dims, strides}. pad: +1 dword every 64 dwords.
__device__ __forceinline__ void tdm_load_k_tile(const float* gptr, unsigned lds_byte_addr) {
    unsigned long long ga = (unsigned long long)(size_t)gptr;
    u32x4 g0;
    g0[0] = 1u;                                         // count=1 (valid user descriptor)
    g0[1] = lds_byte_addr;                              // lds_addr
    g0[2] = (unsigned)(ga & 0xffffffffu);               // global_addr[31:0]
    g0[3] = (unsigned)((ga >> 32) & 0x01ffffffu)        // global_addr[56:32]
          | (2u << 30);                                 // type = 2 ("image")
    i32x8 g1;
    g1[0] = (int)((2u << 16)                            // data_size = 4 bytes
                | (1u << 20)                            // pad_enable
                | (5u << 22)                            // pad_interval: every 64 dwords
                | (0u << 25));                          // pad_amount: 1 dword
    g1[1] = (int)((unsigned)D_ << 16);                  // tensor_dim0 (low16) = 64
    g1[2] = (int)((unsigned)S_ << 16);                  // dim0 hi=0 | tensor_dim1 (low16) = 2048
    g1[3] = (int)((unsigned)D_ << 16);                  // dim1 hi=0 | tile_dim0 = 64
    g1[4] = TILE_M;                                     // tile_dim1 = 16 rows | tile_dim2 = 0
    g1[5] = D_;                                         // tensor_dim0_stride = 64 elements
    g1[6] = 0;
    g1[7] = 0;
    i32x4 z4 = {0, 0, 0, 0};                            // groups 2/3 unused (2D tensor)
    i32x8 z8 = {0, 0, 0, 0, 0, 0, 0, 0};                // trailing group (6-arg toolchain form)
    __builtin_amdgcn_tensor_load_to_lds(g0, g1, z4, z4, z8, 0);
}

__global__ __launch_bounds__(WAVES_PER_BLOCK * 32)
void attn_fused_kernel(const float* __restrict__ Q,
                       const float* __restrict__ K,
                       const float* __restrict__ V,
                       const int*   __restrict__ scale_p,
                       float* __restrict__ Out,
                       float* __restrict__ Att)
{
    extern __shared__ float smem[];

    const int lane = threadIdx.x & 31;
    const int wave = threadIdx.x >> 5;
    const int tile = blockIdx.x * WAVES_PER_BLOCK + wave;   // 0 .. B*H*(S/16)-1
    const int bh   = tile >> 7;                             // tile / (S/16)
    const int m0   = (tile & 127) << 4;                     // query row base

    float* sc   = smem + (size_t)wave * SC_FLOATS;                         // 16 x 2048 scores
    float* sinv = smem + (size_t)WAVES_PER_BLOCK * SC_FLOATS + wave * TILE_M;
    float* kst  = smem + (size_t)WAVES_PER_BLOCK * SC_FLOATS + WAVES_PER_BLOCK * TILE_M
                       + (size_t)wave * (NKBUF * KBUF_FLOATS);             // K staging (2 bufs)
    const unsigned kst_base = (unsigned)(size_t)(void*)kst;                // LDS byte address

    const float inv_scale = 1.0f / (float)scale_p[0];
    const int half = lane >> 4;     // 0: lanes 0-15, 1: lanes 16-31
    const int ln   = lane & 15;

    const float* Qb = Q + ((size_t)bh * S_ + m0) * D_;
    const float* Kb = K + (size_t)bh * S_ * D_;
    const float* Vb = V + (size_t)bh * S_ * D_;

    // ---------- Q tile -> two f16 A fragments (16x32 each, K=0..31 and K=32..63) ----------
    // A fragment, this kernel's contiguous k order (same as B): lane holds row m=ln; element h16 ->
    //   kk = half*16 + h16
    v16h aQ0, aQ1;
    {
        const float* qrow = Qb + (size_t)ln * D_;
#pragma unroll
        for (int g = 0; g < 2; ++g) {
#pragma unroll
            for (int p = 0; p < 8; ++p) {
                const int kk = half * 16 + g * 8 + p;
                aQ0[g * 8 + p] = (_Float16)qrow[kk];
                aQ1[g * 8 + p] = (_Float16)qrow[kk + 32];
            }
        }
    }

    // ---------- pass 1: scores = (Q K^T) * inv_scale -> LDS; K tiles via TDM double buffer ----
    tdm_load_k_tile(Kb, kst_base);                         // prime buffer 0
    for (int j = 0; j < S_ / 16; ++j) {
        if (j + 1 < S_ / 16) {
            tdm_load_k_tile(Kb + (size_t)(j + 1) * 16 * D_,
                            kst_base + ((unsigned)((j + 1) & 1)) * (KBUF_FLOATS * 4u));
            __builtin_amdgcn_s_wait_tensorcnt(1);          // tile j has landed (in-order)
        } else {
            __builtin_amdgcn_s_wait_tensorcnt(0);
        }
        const float* stg = kst + (size_t)(j & 1) * KBUF_FLOATS;

        // B fragment (contiguous k, same order as A): lane holds col n=ln; element h16 -> kk = half*16 + h16.
        const float* krow = stg + (size_t)ln * KSTRIDE + half * 16;
        v16h b0, b1;
#pragma unroll
        for (int p = 0; p < 16; ++p) {
            b0[p] = (_Float16)krow[p];        // d = half*16 + p
            b1[p] = (_Float16)krow[p + 32];   // d = 32 + half*16 + p
        }

        v8f c = {};
        c = __builtin_amdgcn_wmma_f32_16x16x32_f16(false, aQ0, false, b0, (short)0, c, false, false);
        c = __builtin_amdgcn_wmma_f32_16x16x32_f16(false, aQ1, false, b1, (short)0, c, false, false);

        // C layout: VGPR r, lanes 0-15 -> row r, col ln ; lanes 16-31 -> row 8+r, col ln
#pragma unroll
        for (int r = 0; r < 8; ++r)
            sc[(half * 8 + r) * S_ + j * 16 + ln] = c[r] * inv_scale;
    }

    // ---------- softmax: row max ----------
    float rmax[8], rsum[8], rinv[8];
#pragma unroll
    for (int r = 0; r < 8; ++r) rmax[r] = -3.0e38f;
    for (int jj = 0; jj < S_ / 16; ++jj) {
        const int col = jj * 16 + ln;
#pragma unroll
        for (int r = 0; r < 8; ++r)
            rmax[r] = fmaxf(rmax[r], sc[(half * 8 + r) * S_ + col]);
    }
#pragma unroll
    for (int r = 0; r < 8; ++r) {
        float v = rmax[r];
#pragma unroll
        for (int off = 1; off < 16; off <<= 1)
            v = fmaxf(v, __shfl_xor(v, off, 32));
        rmax[r] = v;   // uniform across each 16-lane half
    }

    // ---------- softmax: exp + row sum (exp written back to LDS) ----------
#pragma unroll
    for (int r = 0; r < 8; ++r) rsum[r] = 0.0f;
    for (int jj = 0; jj < S_ / 16; ++jj) {
        const int col = jj * 16 + ln;
#pragma unroll
        for (int r = 0; r < 8; ++r) {
            const int idx = (half * 8 + r) * S_ + col;
            const float e = __expf(sc[idx] - rmax[r]);
            sc[idx] = e;
            rsum[r] += e;
        }
    }
#pragma unroll
    for (int r = 0; r < 8; ++r) {
        float v = rsum[r];
#pragma unroll
        for (int off = 1; off < 16; off <<= 1)
            v += __shfl_xor(v, off, 32);
        rinv[r] = 1.0f / v;
    }
    if (ln == 0) {
#pragma unroll
        for (int r = 0; r < 8; ++r) sinv[half * 8 + r] = rinv[r];
    }
    // cross-lane LDS handoff of row sums (same wave; DS in-order, belt & suspenders)
    asm volatile("s_wait_dscnt 0x0" ::: "memory");

    // ---------- write normalized attention: flat 128KB tile, b128 loads/stores ----------
    // rows m0..m0+15 are contiguous in Att, mirroring the LDS score buffer exactly.
    float* attRow = Att + ((size_t)bh * S_ + m0) * S_;
    for (int row = 0; row < TILE_M; ++row) {
        const float s = sinv[row];                  // wave-uniform scale per row
        const float* src = sc + (size_t)row * S_;
        float*       dst = attRow + (size_t)row * S_;
        for (int i = 0; i < S_ / 128; ++i) {        // 32 lanes x float4 = 128 floats/iter
            const int f = i * 128 + lane * 4;
            float4 v = *(const float4*)(src + f);   // ds_load_b128
            v.x *= s; v.y *= s; v.z *= s; v.w *= s;
            *(float4*)(dst + f) = v;                // global_store_b128, 512B/wave contiguous
        }
    }

    // ---------- P @ V ----------
    v8f o[4] = {};                 // 16x64 output tile = 4 column groups of 16
    const float inv_my = sinv[ln]; // inv row-sum for A-fragment row m = ln
    for (int jc = 0; jc < S_ / 32; ++jc) {
        // P fragment (A layout, row m=ln, kk = half*16 + g*8 + p), gathered from LDS
        v16h aP;
        const float* pr = sc + (size_t)ln * S_ + jc * 32 + half * 16;
#pragma unroll
        for (int g = 0; g < 2; ++g)
#pragma unroll
            for (int p = 0; p < 8; ++p)
                aP[g * 8 + p] = (_Float16)(pr[g * 8 + p] * inv_my);

        if (jc + 1 < S_ / 32)
            __builtin_prefetch(Vb + (size_t)((jc + 1) * 32 + lane) * D_, 0, 2);

#pragma unroll
        for (int nt = 0; nt < 4; ++nt) {
            // V fragment (B layout): element h16 -> kk = half*16 + h16, col n = ln
            const float* vcol = Vb + (size_t)(jc * 32 + half * 16) * D_ + nt * 16 + ln;
            v16h bV;
#pragma unroll
            for (int p = 0; p < 16; ++p)
                bV[p] = (_Float16)vcol[(size_t)p * D_];
            o[nt] = __builtin_amdgcn_wmma_f32_16x16x32_f16(false, aP, false, bV, (short)0, o[nt], false, false);
        }
    }

    // ---------- write output tile ----------
    float* orow = Out + ((size_t)bh * S_ + m0) * D_;
#pragma unroll
    for (int nt = 0; nt < 4; ++nt)
#pragma unroll
        for (int r = 0; r < 8; ++r)
            orow[(size_t)(half * 8 + r) * D_ + nt * 16 + ln] = o[nt][r];
}

extern "C" void kernel_launch(void* const* d_in, const int* in_sizes, int n_in,
                              void* d_out, int out_size, void* d_ws, size_t ws_size,
                              hipStream_t stream) {
    (void)in_sizes; (void)n_in; (void)out_size; (void)d_ws; (void)ws_size;

    const float* Q     = (const float*)d_in[0];
    const float* K     = (const float*)d_in[1];
    const float* V     = (const float*)d_in[2];
    const int*   scale = (const int*)d_in[3];

    float* out = (float*)d_out;                       // [B,H,S,D]
    float* att = out + (size_t)B_ * H_ * S_ * D_;     // [B,H,S,S]

    const int tiles  = B_ * H_ * (S_ / TILE_M);       // 4096 waves of work
    const int blocks = tiles / WAVES_PER_BLOCK;
    const size_t shmem = ((size_t)WAVES_PER_BLOCK * SC_FLOATS
                        + (size_t)WAVES_PER_BLOCK * TILE_M
                        + (size_t)WAVES_PER_BLOCK * NKBUF * KBUF_FLOATS) * sizeof(float);

    attn_fused_kernel<<<blocks, WAVES_PER_BLOCK * 32, shmem, stream>>>(Q, K, V, scale, out, att);
}
